// CLUTRRTransformerModel_25683904430431
// MI455X (gfx1250) — matrix operand order repaired and checked by reading, not compiled, not run on hardware
//
#include <hip/hip_runtime.h>

typedef __bf16 bf16_t;
typedef __attribute__((ext_vector_type(16))) __bf16 v16bf;
typedef __attribute__((ext_vector_type(8)))  __bf16 v8bf;
typedef __attribute__((ext_vector_type(8)))  float  v8f;
typedef __attribute__((ext_vector_type(4)))  unsigned int v4u;
typedef __attribute__((ext_vector_type(8)))  int v8i;
typedef __attribute__((ext_vector_type(4)))  int v4i;

union V16U { v16bf v; v8bf h[2]; };

// model dims
constexpr int cB = 64, cS = 512, cH = 256, cL = 4, cNH = 8, cF = 1024, cK = 8, cR = 25;
constexpr int cDH = cH / cNH;      // 32
constexpr int cM  = cB * cS;       // 32768
constexpr int c3H = 3 * cH;        // 768

__device__ __forceinline__ bf16_t f2bf(float f) {
  unsigned u = __builtin_bit_cast(unsigned, f);
  unsigned short s = (unsigned short)((u + 0x7FFFu + ((u >> 16) & 1u)) >> 16);
  return __builtin_bit_cast(bf16_t, s);
}

// ---------- embedding gather ----------
__global__ void embed_kernel(const int* __restrict__ ids, const float* __restrict__ emb,
                             float* __restrict__ xf, bf16_t* __restrict__ xb) {
  int id = blockIdx.x * blockDim.x + threadIdx.x;
  if (id >= cM * cH) return;
  int hcol = id & (cH - 1);
  int tok  = id >> 8;
  float val = emb[(size_t)ids[tok] * cH + hcol];
  xf[id] = val;
  xb[id] = f2bf(val);
}

// ---------- weight pre-swizzle into WMMA B-operand lane layout (bf16) ----------
// packed[((kc*NT + nt)*32 + lane)*16 + e] = W[kc*32 + 16*(lane>>4) + e][nt*16 + (lane&15)]
__global__ void pack_w_kernel(const float* __restrict__ W, bf16_t* __restrict__ out,
                              int K, int N, int NT) {
  int idx = blockIdx.x * blockDim.x + threadIdx.x;
  int total = (K >> 5) * NT * 512;
  if (idx >= total) return;
  int e    = idx & 15;
  int lane = (idx >> 4) & 31;
  int t    = idx >> 9;
  int nt   = t % NT;
  int kc   = t / NT;
  int k = kc * 32 + 16 * (lane >> 4) + e;
  int n = nt * 16 + (lane & 15);
  float v = (n < N) ? W[(size_t)k * N + n] : 0.f;
  out[idx] = f2bf(v);
}

// ---------- bf16 WMMA GEMM, 2 N-tiles per wave (A-register reuse) ----------
template <bool OBF, bool RELU>
__device__ __forceinline__ void store_tile(void* Cv, const float* bias, v8f acc,
                                           int tm, int tn, int N, int khalf, int lcol) {
  int n = tn * 16 + lcol;
  if (n >= N) return;
  float bia = bias ? bias[n] : 0.f;
#pragma unroll
  for (int r = 0; r < 8; ++r) {
    int row = tm * 16 + 8 * khalf + r;
    float v = acc[r] + bia;
    if (RELU) v = fmaxf(v, 0.f);
    if (OBF) ((bf16_t*)Cv)[(size_t)row * N + n] = f2bf(v);
    else     ((float*)Cv)[(size_t)row * N + n] = v;
  }
}

template <bool OBF, bool RELU>
__global__ void __launch_bounds__(256) gemm_wmma(const bf16_t* __restrict__ A,
                                                 const bf16_t* __restrict__ Bp,
                                                 const float* __restrict__ bias,
                                                 void* __restrict__ Cv,
                                                 int M, int N, int K) {
  int lane = threadIdx.x & 31;
  int wid  = blockIdx.x * (blockDim.x >> 5) + (threadIdx.x >> 5);
  int tiles_n = (N + 15) >> 4;
  int tnp     = (tiles_n + 1) >> 1;      // pairs of n-tiles per wave
  int tiles   = (M >> 4) * tnp;
  if (wid >= tiles) return;
  int tm = wid / tnp, tp = wid - tm * tnp;
  int tn0 = tp * 2;
  bool has2 = (tn0 + 1) < tiles_n;
  int khalf = lane >> 4, lcol = lane & 15;

  const bf16_t* arow = A + (size_t)(tm * 16 + lcol) * K + 16 * khalf;
  size_t bstride = (size_t)tiles_n * 512;
  const bf16_t* bp0 = Bp + ((size_t)tn0 * 32 + lane) * 16;
  const bf16_t* bp1 = bp0 + 512;

  v8f acc0 = {}, acc1 = {};
  int KC = K >> 5;
  for (int kc = 0; kc < KC; ++kc) {
    V16U a;
    a.h[0] = *(const v8bf*)(arow + kc * 32);       // K = 16*khalf + [0,8)
    a.h[1] = *(const v8bf*)(arow + kc * 32 + 8);   // K = 16*khalf + 8 + [0,8)
    v16bf b0 = *(const v16bf*)(bp0 + (size_t)kc * bstride);
    acc0 = __builtin_amdgcn_wmma_f32_16x16x32_bf16(false, a.v, false, b0,
                                                   (short)0, acc0, false, false);
    if (has2) {
      v16bf b1 = *(const v16bf*)(bp1 + (size_t)kc * bstride);
      acc1 = __builtin_amdgcn_wmma_f32_16x16x32_bf16(false, a.v, false, b1,
                                                     (short)0, acc1, false, false);
    }
  }
  store_tile<OBF, RELU>(Cv, bias, acc0, tm, tn0, N, khalf, lcol);
  if (has2) store_tile<OBF, RELU>(Cv, bias, acc1, tm, tn0 + 1, N, khalf, lcol);
}

// ---------- attention: one workgroup per (b, head, 16-query tile) ----------
// V slab (512 keys x 32 dh, row stride 768 elems) staged to LDS by the Tensor
// Data Mover, overlapped with the QK^T + softmax phase. TDM pads 1 DWORD per
// 16 DWORDs (per 64B row) -> LDS row stride 68B, rotating banks every 16 keys.
__global__ void __launch_bounds__(256) attn_kernel(const bf16_t* __restrict__ qkv,
                                                   const int* __restrict__ amask,
                                                   bf16_t* __restrict__ out) {
  __shared__ float  s_sc[16][cS + 8];
  __shared__ bf16_t s_p[16][cS + 8];
  __shared__ float  s_out[16][cDH];
  __shared__ bf16_t s_v[cS][cDH + 2];   // 68B row stride (TDM pad)
  int tid = threadIdx.x;
  int lane = tid & 31, wave = tid >> 5;
  int khalf = lane >> 4, lcol = lane & 15;
  int qt = blockIdx.x & 31;
  int h  = (blockIdx.x >> 5) & 7;
  int b  = blockIdx.x >> 8;
  const float scale = 0.17677669529663687f;  // 1/sqrt(DH=32)

  // ---- issue async TDM load of the V slab (wave 0 only) ----
  if (wave == 0) {
    unsigned long long gaddr = (unsigned long long)(const void*)qkv
        + 2ull * ((size_t)(b * cS) * c3H + 2 * cH + h * cDH);
    unsigned ldsa = (unsigned)(size_t)&s_v[0][0];
    v4u g0;
    g0.x = 1u;                                      // count=1 valid descriptor
    g0.y = ldsa;                                    // lds_addr
    g0.z = (unsigned)(gaddr & 0xFFFFFFFFu);         // global_addr[31:0]
    g0.w = (unsigned)((gaddr >> 32) & 0x1FFFFFFu)   // global_addr[56:32]
         | (2u << 30);                              // type=2 (image)
    v8i g1;
    g1[0] = (1 << 16)        // data_size = 2 bytes
          | (1 << 20)        // pad_enable
          | (3 << 22);       // pad_interval: 16 DWORDs (one 64B row)
                             // pad_amount field 0 -> 1 DWORD
    g1[1] = (cDH << 16);                 // tensor_dim0 = 32 (lo 16 in bits 31:16)
    g1[2] = (cS << 16);                  // tensor_dim0 hi=0 | tensor_dim1 = 512
    g1[3] = (16 << 16) * 2;              // tensor_dim1 hi=0 | tile_dim0 = 32
    g1[4] = cS;                          // tile_dim1 = 512, tile_dim2 = 0
    g1[5] = c3H;                         // tensor_dim0_stride = 768 (lo 32)
    g1[6] = 0;                           // stride0 hi | tensor_dim1_stride lo
    g1[7] = 0;
    v4i gz = {0, 0, 0, 0};
    v8i gz8 = {0, 0, 0, 0, 0, 0, 0, 0};
    __builtin_amdgcn_tensor_load_to_lds(g0, g1, gz, gz, gz8, 0);
  }

  // ---- Q tile A-operand (16 queries x DH=32: exactly one WMMA K-slab) ----
  const bf16_t* qrow = qkv + (size_t)(b * cS + qt * 16 + lcol) * c3H + h * cDH;
  V16U aq;
  aq.h[0] = *(const v8bf*)(qrow + 16 * khalf);
  aq.h[1] = *(const v8bf*)(qrow + 16 * khalf + 8);

  // ---- scores: 32 key tiles, 4 per wave; K^T B-operand contiguous per lane ----
  for (int kt = wave * 4; kt < wave * 4 + 4; ++kt) {
    int key = kt * 16 + lcol;
    const bf16_t* krow = qkv + (size_t)(b * cS + key) * c3H + cH + h * cDH + 16 * khalf;
    V16U bk;
    bk.h[0] = *(const v8bf*)(krow);
    bk.h[1] = *(const v8bf*)(krow + 8);
    v8f sc = {};
    sc = __builtin_amdgcn_wmma_f32_16x16x32_bf16(false, aq.v, false, bk.v,
                                                 (short)0, sc, false, false);
    float kb = (amask[b * cS + key] > 0) ? 0.f : -1e9f;
#pragma unroll
    for (int r = 0; r < 8; ++r) s_sc[8 * khalf + r][key] = sc[r] * scale + kb;
  }
  for (int i = tid; i < 16 * cDH; i += 256) ((float*)s_out)[i] = 0.f;
  __syncthreads();

  // ---- softmax: wave w owns rows 2w (lanes 0-15) and 2w+1 (lanes 16-31) ----
  {
    int row = 2 * wave + khalf;
    float mx = -1e30f;
#pragma unroll
    for (int j = 0; j < 32; ++j) mx = fmaxf(mx, s_sc[row][lcol + 16 * j]);
    for (int o = 8; o; o >>= 1) mx = fmaxf(mx, __shfl_xor(mx, o));
    float ev[32], sum = 0.f;
#pragma unroll
    for (int j = 0; j < 32; ++j) { ev[j] = __expf(s_sc[row][lcol + 16 * j] - mx); sum += ev[j]; }
    for (int o = 8; o; o >>= 1) sum += __shfl_xor(sum, o);
    float inv = 1.f / sum;
#pragma unroll
    for (int j = 0; j < 32; ++j) s_p[row][lcol + 16 * j] = f2bf(ev[j] * inv);
  }
  if (wave == 0) __builtin_amdgcn_s_wait_tensorcnt(0);   // V slab resident
  __syncthreads();

  // ---- P @ V with split-K across waves; V B-operand gathered from LDS ----
#pragma unroll
  for (int nt = 0; nt < 2; ++nt) {
    v8f acc = {};
#pragma unroll
    for (int kk = 0; kk < 2; ++kk) {
      int kc = wave * 2 + kk;
      V16U ap;
      ap.h[0] = *(const v8bf*)&s_p[lcol][kc * 32 + 16 * khalf];
      ap.h[1] = *(const v8bf*)&s_p[lcol][kc * 32 + 16 * khalf + 8];
      int dh = nt * 16 + lcol;
      v16bf bv;
#pragma unroll
      for (int e = 0; e < 16; ++e) bv[e] = s_v[kc * 32 + 16 * khalf + e][dh];
      acc = __builtin_amdgcn_wmma_f32_16x16x32_bf16(false, ap.v, false, bv,
                                                    (short)0, acc, false, false);
    }
#pragma unroll
    for (int r = 0; r < 8; ++r)
      atomicAdd(&s_out[8 * khalf + r][nt * 16 + lcol], acc[r]);
  }
  __syncthreads();

  for (int i = tid; i < 16 * cDH; i += 256) {
    int row = i >> 5, d = i & 31;
    out[(size_t)(b * cS + qt * 16 + row) * cH + h * cDH + d] = f2bf(s_out[row][d]);
  }
}

// ---------- residual add + LayerNorm (delta==nullptr -> plain LN) ----------
__global__ void resid_ln_kernel(float* __restrict__ xf, const float* __restrict__ delta,
                                const float* __restrict__ g, const float* __restrict__ bta,
                                bf16_t* __restrict__ xb) {
  __shared__ float part[8];
  int row = blockIdx.x, tid = threadIdx.x;
  size_t off = (size_t)row * cH + tid;
  float v = xf[off];
  if (delta) v += delta[off];
  float s = v;
  for (int o = 16; o; o >>= 1) s += __shfl_xor(s, o);
  if ((tid & 31) == 0) part[tid >> 5] = s;
  __syncthreads();
  float mu = 0.f;
#pragma unroll
  for (int i = 0; i < 8; ++i) mu += part[i];
  mu *= (1.f / cH);
  float d = v - mu;
  float q = d * d;
  for (int o = 16; o; o >>= 1) q += __shfl_xor(q, o);
  __syncthreads();
  if ((tid & 31) == 0) part[tid >> 5] = q;
  __syncthreads();
  float var = 0.f;
#pragma unroll
  for (int i = 0; i < 8; ++i) var += part[i];
  var *= (1.f / cH);
  float y = d * rsqrtf(var + 1e-5f) * g[tid] + bta[tid];
  xf[off] = y;
  if (xb) xb[off] = f2bf(y);
}

// ---------- pooled sub/obj + masked story mean -> bf16 feature [B, 3H] ----------
__global__ void pool_kernel(const float* __restrict__ seq, const int* __restrict__ amask,
                            const int* __restrict__ sidx, const int* __restrict__ scnt,
                            const int* __restrict__ oidx, const int* __restrict__ ocnt,
                            bf16_t* __restrict__ featb) {
  int b = blockIdx.x, hcol = threadIdx.x;
  const float* sb = seq + (size_t)b * cS * cH;
  float ssub = 0.f; int nc = scnt[b];
  for (int k = 0; k < cK; ++k)
    if (k < nc) ssub += sb[(size_t)sidx[b * cK + k] * cH + hcol];
  float subv = nc > 0 ? ssub / (float)nc : 0.f;
  float sobj = 0.f; int nco = ocnt[b];
  for (int k = 0; k < cK; ++k)
    if (k < nco) sobj += sb[(size_t)oidx[b * cK + k] * cH + hcol];
  float objv = nco > 0 ? sobj / (float)nco : 0.f;
  float acc = 0.f, len = 0.f;
  for (int s = 0; s < cS; ++s)
    if (amask[b * cS + s] > 0) { acc += sb[(size_t)s * cH + hcol]; len += 1.f; }
  float story = acc / fmaxf(len, 1.f);
  size_t fb = (size_t)b * c3H;
  featb[fb + hcol]          = f2bf(subv);
  featb[fb + cH + hcol]     = f2bf(objv);
  featb[fb + 2 * cH + hcol] = f2bf(story);
}

extern "C" void kernel_launch(void* const* d_in, const int* in_sizes, int n_in,
                              void* d_out, int out_size, void* d_ws, size_t ws_size,
                              hipStream_t stream) {
  (void)in_sizes; (void)n_in; (void)out_size; (void)ws_size;
  const int*   ids     = (const int*)d_in[0];
  const int*   amask   = (const int*)d_in[1];
  const int*   sub_idx = (const int*)d_in[2];
  const int*   sub_cnt = (const int*)d_in[3];
  const int*   obj_idx = (const int*)d_in[4];
  const int*   obj_cnt = (const int*)d_in[5];
  const float* emb     = (const float*)d_in[6];
  const float* Wqkv    = (const float*)d_in[7];
  const float* bqkv    = (const float*)d_in[8];
  const float* Wo      = (const float*)d_in[9];
  const float* bo      = (const float*)d_in[10];
  const float* ln1g    = (const float*)d_in[11];
  const float* ln1b    = (const float*)d_in[12];
  const float* ln2g    = (const float*)d_in[13];
  const float* ln2b    = (const float*)d_in[14];
  const float* W1      = (const float*)d_in[15];
  const float* b1      = (const float*)d_in[16];
  const float* W2      = (const float*)d_in[17];
  const float* b2      = (const float*)d_in[18];
  const float* lnfg    = (const float*)d_in[19];
  const float* lnfb    = (const float*)d_in[20];
  const float* mW1     = (const float*)d_in[21];
  const float* mb1     = (const float*)d_in[22];
  const float* mW2     = (const float*)d_in[23];
  const float* mb2     = (const float*)d_in[24];

  char* pws = (char*)d_ws;
  auto bump = [&](size_t n) { char* r = pws; pws += (n + 255) & ~(size_t)255; return r; };
  float*  xf    = (float*)bump((size_t)cM * cH * 4);
  bf16_t* xb    = (bf16_t*)bump((size_t)cM * cH * 2);
  bf16_t* qkvb  = (bf16_t*)bump((size_t)cM * c3H * 2);
  bf16_t* aob   = (bf16_t*)bump((size_t)cM * cH * 2);
  float*  d1f   = (float*)bump((size_t)cM * cH * 4);
  bf16_t* hb    = (bf16_t*)bump((size_t)cM * cF * 2);
  bf16_t* pWqkv = (bf16_t*)bump((size_t)cL * 196608 * 2);
  bf16_t* pWo   = (bf16_t*)bump((size_t)cL * 65536 * 2);
  bf16_t* pW1   = (bf16_t*)bump((size_t)cL * 262144 * 2);
  bf16_t* pW2   = (bf16_t*)bump((size_t)cL * 262144 * 2);
  bf16_t* pmW1  = (bf16_t*)bump((size_t)196608 * 2);
  bf16_t* pmW2  = (bf16_t*)bump((size_t)8192 * 2);
  bf16_t* featb = (bf16_t*)bump((size_t)cB * c3H * 2);
  bf16_t* headb = (bf16_t*)bump((size_t)cB * cH * 2);

  auto pack = [&](const float* W, bf16_t* o, int K, int N) {
    int NT = (N + 15) / 16;
    int total = (K / 32) * NT * 512;
    pack_w_kernel<<<(total + 255) / 256, 256, 0, stream>>>(W, o, K, N, NT);
  };
  for (int l = 0; l < cL; ++l) {
    pack(Wqkv + (size_t)l * cH * c3H, pWqkv + (size_t)l * 196608, cH, c3H);
    pack(Wo   + (size_t)l * cH * cH,  pWo   + (size_t)l * 65536,  cH, cH);
    pack(W1   + (size_t)l * cH * cF,  pW1   + (size_t)l * 262144, cH, cF);
    pack(W2   + (size_t)l * cF * cH,  pW2   + (size_t)l * 262144, cF, cH);
  }
  pack(mW1, pmW1, c3H, cH);
  pack(mW2, pmW2, cH, cR);

  embed_kernel<<<(cM * cH) / 256, 256, 0, stream>>>(ids, emb, xf, xb);

  auto gb = [](int M, int N) {
    int tn = (N + 15) / 16;
    int t = (M / 16) * ((tn + 1) / 2);
    return (t + 7) / 8;
  };

  for (int l = 0; l < cL; ++l) {
    gemm_wmma<true, false><<<gb(cM, c3H), 256, 0, stream>>>(
        xb, pWqkv + (size_t)l * 196608, bqkv + l * c3H, qkvb, cM, c3H, cH);
    attn_kernel<<<cB * cNH * (cS / 16), 256, 0, stream>>>(qkvb, amask, aob);
    gemm_wmma<false, false><<<gb(cM, cH), 256, 0, stream>>>(
        aob, pWo + (size_t)l * 65536, bo + l * cH, d1f, cM, cH, cH);
    resid_ln_kernel<<<cM, cH, 0, stream>>>(xf, d1f, ln1g + l * cH, ln1b + l * cH, xb);
    gemm_wmma<true, true><<<gb(cM, cF), 256, 0, stream>>>(
        xb, pW1 + (size_t)l * 262144, b1 + l * cF, hb, cM, cF, cH);
    gemm_wmma<false, false><<<gb(cM, cH), 256, 0, stream>>>(
        hb, pW2 + (size_t)l * 262144, b2 + l * cH, d1f, cM, cH, cF);
    resid_ln_kernel<<<cM, cH, 0, stream>>>(xf, d1f, ln2g + l * cH, ln2b + l * cH, xb);
  }
  resid_ln_kernel<<<cM, cH, 0, stream>>>(xf, nullptr, lnfg, lnfb, nullptr);
  pool_kernel<<<cB, cH, 0, stream>>>(xf, amask, sub_idx, sub_cnt, obj_idx, obj_cnt, featb);
  gemm_wmma<true, true><<<gb(cB, cH), 256, 0, stream>>>(featb, pmW1, mb1, headb, cB, cH, c3H);
  gemm_wmma<false, false><<<gb(cB, cR), 256, 0, stream>>>(headb, pmW2, mb2, d_out, cB, cR, cH);
}
